// SwinTransformerBlock_5471788335758
// MI455X (gfx1250) — matrix operand order repaired and checked by reading, not compiled, not run on hardware
//
#include <hip/hip_runtime.h>

#define B_   4
#define N_   16384
#define C_   512
#define H_   16
#define D_   32
#define HID_ 2048
#define M_   (B_ * N_)      // 65536 tokens

typedef __attribute__((ext_vector_type(8)))  float   v8f;
typedef __bf16 bf16_t;
typedef __attribute__((ext_vector_type(8)))  bf16_t  v8bf;
typedef __attribute__((ext_vector_type(16))) bf16_t  v16bf;

// ---------------- helpers ----------------

__device__ __forceinline__ unsigned short f2bf(float f) {
  unsigned u = __float_as_uint(f);
  u += 0x7FFFu + ((u >> 16) & 1u);          // round-to-nearest-even
  return (unsigned short)(u >> 16);
}
__device__ __forceinline__ float bf2f(unsigned short h) {
  return __uint_as_float(((unsigned)h) << 16);
}
// generic LDS pointer: low 32 bits are the LDS byte offset
__device__ __forceinline__ unsigned lds_off(const void* p) {
  return (unsigned)(unsigned long long)p;
}
// CDNA5 async copy global -> LDS (16B per lane), tracked by ASYNCcnt
__device__ __forceinline__ void async_ld_b128(unsigned dstLds, const void* src) {
  asm volatile("global_load_async_to_lds_b128 %0, %1, off"
               :: "v"(dstLds), "v"((unsigned long long)src)
               : "memory");
}
__device__ __forceinline__ void wait_async0() {
#if __has_builtin(__builtin_amdgcn_s_wait_asynccnt)
  __builtin_amdgcn_s_wait_asynccnt(0);
#else
  asm volatile("s_wait_asynccnt 0" ::: "memory");
#endif
}
__device__ __forceinline__ void wait_async4() {
#if __has_builtin(__builtin_amdgcn_s_wait_asynccnt)
  __builtin_amdgcn_s_wait_asynccnt(4);
#else
  asm volatile("s_wait_asynccnt 4" ::: "memory");
#endif
}
// build a v16bf fragment from two 16B LDS chunks (ds_load_b128 x2)
__device__ __forceinline__ v16bf ld_frag16(const unsigned short* p, int off2) {
  v8bf lo = *reinterpret_cast<const v8bf*>(p);
  v8bf hi = *reinterpret_cast<const v8bf*>(p + off2);
  return __builtin_shufflevector(lo, hi, 0,1,2,3,4,5,6,7,8,9,10,11,12,13,14,15);
}
// branch-free tanh-form GELU: v * sigmoid(2y) = v * rcp(1 + exp2(-2.3025851*v*(1+0.044715 v^2)))
__device__ __forceinline__ float gelu_fast(float v) {
  const float t = -2.302585093f * v * (1.f + 0.044715f * v * v);
  const float e = __builtin_amdgcn_exp2f(t);
  return v * __builtin_amdgcn_rcpf(1.f + e);
}

// ---------------- 1) fp32 -> bf16 weight conversion ----------------

__global__ __launch_bounds__(256) void cvt_f32_bf16(const float* __restrict__ s,
                                                    unsigned short* __restrict__ d) {
  const int i = (blockIdx.x * 256 + threadIdx.x) * 4;
  const float4 f = *reinterpret_cast<const float4*>(s + i);
  uint2 r;
  r.x = (unsigned)f2bf(f.x) | ((unsigned)f2bf(f.y) << 16);
  r.y = (unsigned)f2bf(f.z) | ((unsigned)f2bf(f.w) << 16);
  *reinterpret_cast<uint2*>(d + i) = r;
}

// -------- 2) LN1 + per-head LN_std -> h(f32), q/k/v(bf16) --------

__global__ __launch_bounds__(256) void ln1_qkv_kernel(
    const float* __restrict__ x,
    const float* __restrict__ w,  const float* __restrict__ b,
    const float* __restrict__ kw, const float* __restrict__ kb,
    const float* __restrict__ vw, const float* __restrict__ vb,
    float* __restrict__ hout, unsigned short* __restrict__ qbf,
    unsigned short* __restrict__ kbf, unsigned short* __restrict__ vbf)
{
  const int lane  = threadIdx.x & 31;
  const int token = blockIdx.x * 8 + (threadIdx.x >> 5);
  const size_t base = (size_t)token * C_ + lane * 16;

  float xv[16];
  const float4* x4 = reinterpret_cast<const float4*>(x + base);
  float s = 0.f, sq = 0.f;
#pragma unroll
  for (int i = 0; i < 4; ++i) {
    float4 f = x4[i];
    xv[i*4+0]=f.x; xv[i*4+1]=f.y; xv[i*4+2]=f.z; xv[i*4+3]=f.w;
  }
#pragma unroll
  for (int i = 0; i < 16; ++i) { s += xv[i]; sq += xv[i]*xv[i]; }
#pragma unroll
  for (int m = 16; m >= 1; m >>= 1) {
    s  += __shfl_xor(s,  m, 32);
    sq += __shfl_xor(sq, m, 32);
  }
  const float mean = s * (1.f / C_);
  const float rstd = rsqrtf(sq * (1.f / C_) - mean * mean + 1e-5f);

  float hv[16];
  const float* wp = w + lane * 16;
  const float* bp = b + lane * 16;
#pragma unroll
  for (int i = 0; i < 16; ++i) hv[i] = wp[i] * (xv[i] - mean) * rstd + bp[i];

  float4* h4 = reinterpret_cast<float4*>(hout + base);
#pragma unroll
  for (int i = 0; i < 4; ++i) {
    float4 f; f.x=hv[i*4]; f.y=hv[i*4+1]; f.z=hv[i*4+2]; f.w=hv[i*4+3];
    h4[i] = f;
  }
  // bf16 copy of h for the WMMA attention kernel (q == h)
  union { unsigned short u[16]; uint4 q[2]; } qo;
#pragma unroll
  for (int i = 0; i < 16; ++i) qo.u[i] = f2bf(hv[i]);
  *reinterpret_cast<uint4*>(qbf + base)     = qo.q[0];
  *reinterpret_cast<uint4*>(qbf + base + 8) = qo.q[1];

  // per-head LN_std (unbiased var, eps added to std); head spans 2 lanes
  float hs = 0.f, hq = 0.f;
#pragma unroll
  for (int i = 0; i < 16; ++i) { hs += hv[i]; hq += hv[i]*hv[i]; }
  hs += __shfl_xor(hs, 1, 32);
  hq += __shfl_xor(hq, 1, 32);
  const float hm   = hs * (1.f / D_);
  const float varu = (hq - (float)D_ * hm * hm) * (1.f / (D_ - 1));
  const float rden = 1.f / (sqrtf(fmaxf(varu, 0.f)) + 1e-5f);

  const int head  = lane >> 1;
  const int dbase = (lane & 1) * 16;
  const int bi = token >> 14, n = token & (N_ - 1);
  const size_t koff = ((size_t)(bi * H_ + head) * N_ + n) * D_ + dbase;
  const float* kwp = kw + head * D_ + dbase;  const float* kbp = kb + head * D_ + dbase;
  const float* vwp = vw + head * D_ + dbase;  const float* vbp = vb + head * D_ + dbase;

  union { unsigned short u[16]; uint4 q[2]; } ko, vo;
#pragma unroll
  for (int i = 0; i < 16; ++i) {
    const float cn = (hv[i] - hm) * rden;
    ko.u[i] = f2bf(kwp[i] * cn + kbp[i]);
    vo.u[i] = f2bf(vwp[i] * cn + vbp[i]);
  }
  *reinterpret_cast<uint4*>(kbf + koff)     = ko.q[0];
  *reinterpret_cast<uint4*>(kbf + koff + 8) = ko.q[1];
  *reinterpret_cast<uint4*>(vbf + koff)     = vo.q[0];
  *reinterpret_cast<uint4*>(vbf + koff + 8) = vo.q[1];
}

// -------- 3) kv = k^T v / N via WMMA (in-LDS transpose staging) --------
// block: one (b,h), 1024 n-rows in 4 staged passes; each wave owns a 32-n slice.

#define NP_ 264                       // padded n-stride (halves) for kT/vT
#define KVT_BYTES (2 * 32 * NP_ * 2)  // 33,792 B (reduction buffer aliases this)

__global__ __launch_bounds__(256) void kv_accum_kernel(
    const unsigned short* __restrict__ kbf, const unsigned short* __restrict__ vbf,
    float* __restrict__ kv)
{
  __shared__ __align__(16) char smem[KVT_BYTES];   // kT|vT, later reused as red[8][1024]
  unsigned short* kT = reinterpret_cast<unsigned short*>(smem);              // [d][n]
  unsigned short* vT = reinterpret_cast<unsigned short*>(smem + 32*NP_*2);   // [e][n]
  float* red = reinterpret_cast<float*>(smem);

  const int t    = threadIdx.x;
  const int lane = t & 31;
  const int wv   = t >> 5;
  const int bh   = blockIdx.y;
  const int n0   = blockIdx.x * 1024;

  v8f acc[2][2];
#pragma unroll
  for (int i = 0; i < 2; ++i)
#pragma unroll
    for (int j = 0; j < 2; ++j) {
      v8f z = {0.f,0.f,0.f,0.f,0.f,0.f,0.f,0.f};
      acc[i][j] = z;
    }

  const int ar  = lane & 15;
  const int akh = (lane >> 4) * 8;
  const int bkh = (lane >> 4) * 16;
  const int nb  = wv * 32;            // this wave's 32-n slice

  for (int ps = 0; ps < 4; ++ps) {
    if (ps) __syncthreads();          // previous pass fully consumed
    // thread t loads row n = ps*256+t (32 halves) and scatters transposed
    const size_t nrow = (size_t)bh * N_ + n0 + ps * 256 + t;
    const uint4* ks = reinterpret_cast<const uint4*>(kbf + nrow * D_);
    const uint4* vs = reinterpret_cast<const uint4*>(vbf + nrow * D_);
    union { uint4 q[4]; unsigned short u[32]; } kr, vr;
#pragma unroll
    for (int s = 0; s < 4; ++s) { kr.q[s] = ks[s]; vr.q[s] = vs[s]; }
#pragma unroll
    for (int d = 0; d < 32; ++d) {
      kT[d * NP_ + t] = kr.u[d];
      vT[d * NP_ + t] = vr.u[d];
    }
    __syncthreads();

    // A = kT (32 d x 32 n slice), B = vT^T (32 n x 32 e slice): 4 WMMA tiles
    v16bf af[2], bfr[2];
#pragma unroll
    for (int i = 0; i < 2; ++i)
      af[i]  = ld_frag16(&kT[(i*16 + ar) * NP_ + nb + akh], 16);
#pragma unroll
    for (int j = 0; j < 2; ++j)
      bfr[j] = ld_frag16(&vT[(j*16 + ar) * NP_ + nb + akh], 16);
#pragma unroll
    for (int i = 0; i < 2; ++i)
#pragma unroll
      for (int j = 0; j < 2; ++j)
        acc[i][j] = __builtin_amdgcn_wmma_f32_16x16x32_bf16(
            false, af[i], false, bfr[j], (short)0, acc[i][j], false, false);
  }
  __syncthreads();

  // cross-wave reduction: partials into LDS (aliases kT/vT), then fp32 atomics
  float* rw = red + wv * 1024;
#pragma unroll
  for (int i = 0; i < 2; ++i)
#pragma unroll
    for (int j = 0; j < 2; ++j)
#pragma unroll
      for (int r = 0; r < 8; ++r)
        rw[(i*16 + (lane >> 4)*8 + r) * 32 + j*16 + (lane & 15)] = acc[i][j][r];
  __syncthreads();

  const float sc = 1.f / (float)N_;
#pragma unroll
  for (int u = 0; u < 4; ++u) {
    const int idx = u * 256 + t;      // idx = d*32 + e
    float ssum = 0.f;
#pragma unroll
    for (int wvi = 0; wvi < 8; ++wvi) ssum += red[wvi * 1024 + idx];
    atomicAdd(&kv[(size_t)bh * 1024 + idx], ssum * sc);
  }
}

// -------- 4) attn = q*kv via WMMA, x2 = x + h + attn -> d_out --------
// block: 16 tokens x all 512 cols = 32 WMMA tiles; wave wv owns heads {2wv,2wv+1}.

#define QSTR_ 520   // padded halves per staged q row

__global__ __launch_bounds__(256) void attn_resid_kernel(
    const float* __restrict__ x, const float* __restrict__ h,
    const unsigned short* __restrict__ qbf,
    const float* __restrict__ kv, float* __restrict__ out)
{
  __shared__ __align__(16) unsigned short As[16 * QSTR_];     // [token][c] 16,640 B
  __shared__ __align__(16) unsigned short kvt[16 * 32 * 40];  // [head][e][d] 40,960 B
  const int t      = threadIdx.x;
  const int lane   = t & 31;
  const int wv     = t >> 5;
  const int token0 = blockIdx.x * 16;
  const int bi     = token0 >> 14;

  // stage 16 q rows (16 KB) via async DMA
  const unsigned asB = lds_off(&As[0]);
#pragma unroll
  for (int s = 0; s < 4; ++s) {
    const int c = s * 256 + t;              // 1024 chunks of 16B
    const int row = c >> 6, q = c & 63;
    async_ld_b128(asB + row * (QSTR_ * 2) + q * 16,
                  qbf + (size_t)(token0 + row) * C_ + q * 8);
  }
  // meanwhile build kvt[head][e][d] (bf16, pad 40) from fp32 kv (L2-hot)
  const float* kvb = kv + (size_t)bi * (H_ * D_ * D_);
  for (int rr = t; rr < 512; rr += 256) {   // rr = head*32 + d
    const int head = rr >> 5, d = rr & 31;
    const float* src = kvb + rr * 32;
#pragma unroll
    for (int e = 0; e < 32; ++e)
      kvt[(head*32 + e)*40 + d] = f2bf(src[e]);
  }
  wait_async0();
  __syncthreads();

  const int ar  = lane & 15;
  const int akh = (lane >> 4) * 8;
  const int bkh = (lane >> 4) * 16;
  v8f acc[2][2];                            // [head-sub][e-half]
#pragma unroll
  for (int hh = 0; hh < 2; ++hh) {
    const int head = wv * 2 + hh;
    const v16bf af = ld_frag16(&As[ar * QSTR_ + head * 32 + akh], 16);
#pragma unroll
    for (int eh = 0; eh < 2; ++eh) {
      const v16bf bb = ld_frag16(&kvt[(head*32 + eh*16 + ar)*40 + akh], 16);
      v8f z = {0.f,0.f,0.f,0.f,0.f,0.f,0.f,0.f};
      acc[hh][eh] = __builtin_amdgcn_wmma_f32_16x16x32_bf16(
          false, af, false, bb, (short)0, z, false, false);
    }
  }

  // epilogue: out = x + h + attn (fp32)
  const int trow = (lane >> 4) * 8;
#pragma unroll
  for (int hh = 0; hh < 2; ++hh) {
    const int head = wv * 2 + hh;
#pragma unroll
    for (int eh = 0; eh < 2; ++eh) {
      const int c = head*32 + eh*16 + (lane & 15);
#pragma unroll
      for (int r = 0; r < 8; ++r) {
        const size_t idx = (size_t)(token0 + trow + r) * C_ + c;
        out[idx] = x[idx] + h[idx] + acc[hh][eh][r];
      }
    }
  }
}

// ---------------- 5) LN2 -> bf16 m ----------------

__global__ __launch_bounds__(256) void ln2_kernel(
    const float* __restrict__ x2, const float* __restrict__ w, const float* __restrict__ b,
    unsigned short* __restrict__ mbf)
{
  const int lane  = threadIdx.x & 31;
  const int token = blockIdx.x * 8 + (threadIdx.x >> 5);
  const size_t base = (size_t)token * C_ + lane * 16;
  float xv[16];
  const float4* x4 = reinterpret_cast<const float4*>(x2 + base);
  float s = 0.f, sq = 0.f;
#pragma unroll
  for (int i = 0; i < 4; ++i) {
    float4 f = x4[i];
    xv[i*4+0]=f.x; xv[i*4+1]=f.y; xv[i*4+2]=f.z; xv[i*4+3]=f.w;
  }
#pragma unroll
  for (int i = 0; i < 16; ++i) { s += xv[i]; sq += xv[i]*xv[i]; }
#pragma unroll
  for (int m = 16; m >= 1; m >>= 1) {
    s  += __shfl_xor(s,  m, 32);
    sq += __shfl_xor(sq, m, 32);
  }
  const float mean = s * (1.f / C_);
  const float rstd = rsqrtf(sq * (1.f / C_) - mean * mean + 1e-5f);
  const float* wp = w + lane * 16;
  const float* bp = b + lane * 16;
  union { unsigned short u[16]; uint4 q[2]; } mo;
#pragma unroll
  for (int i = 0; i < 16; ++i)
    mo.u[i] = f2bf(wp[i] * (xv[i] - mean) * rstd + bp[i]);
  *reinterpret_cast<uint4*>(mbf + base)     = mo.q[0];
  *reinterpret_cast<uint4*>(mbf + base + 8) = mo.q[1];
}

// ------ 6/7) bf16 WMMA GEMM, double-buffered async LDS pipeline ------
// Out[M,N] = A[M,K] * W[N,K]^T (+bias, then GELU or +resid)
// grid.x = N tiles (fast-varying -> A rows stay hot in cache, W lives in L2)

#define LDA_    40                      // halves per LDS row (32 data + 8 pad)
#define TILEHB  (128 * LDA_)            // halves per buffer
#define TILEB   (TILEHB * 2)            // bytes per buffer

template<bool GELU, bool OUT_BF16>
__global__ __launch_bounds__(256) void gemm_bf16_kernel(
    const unsigned short* __restrict__ A,   // [M,K] bf16
    const unsigned short* __restrict__ W,   // [Ncols,K] bf16
    const float* __restrict__ bias,         // [Ncols]
    const float* __restrict__ resid,        // [M,Ncols] (used when !OUT_BF16)
    void* __restrict__ Out,
    int K, int Ncols)
{
  __shared__ __align__(16) unsigned short As[2 * TILEHB];
  __shared__ __align__(16) unsigned short Ws[2 * TILEHB];

  const int t    = threadIdx.x;
  const int lane = t & 31;
  const int wave = t >> 5;      // 8 waves: 4 x 2
  const int wm   = wave >> 1;   // 0..3  -> 32 rows each
  const int wn   = wave & 1;    // 0..1  -> 64 cols each
  const int bn   = blockIdx.x * 128;   // N tile (fast-varying)
  const int bm   = blockIdx.y * 128;   // M tile

  v8f acc[2][4];
#pragma unroll
  for (int i = 0; i < 2; ++i)
#pragma unroll
    for (int j = 0; j < 4; ++j) {
      v8f z = {0.f,0.f,0.f,0.f,0.f,0.f,0.f,0.f};
      acc[i][j] = z;
    }

  const unsigned asB = lds_off(&As[0]);
  const unsigned wsB = lds_off(&Ws[0]);
  const int nk = K / 32;

  auto stage = [&](int kidx, int bsel) {
#pragma unroll
    for (int s2 = 0; s2 < 2; ++s2) {
      const int c   = t * 2 + s2;
      const int row = c >> 2, q = c & 3;
      const int kk  = kidx * 32 + q * 8;
      async_ld_b128(asB + bsel * TILEB + row * (LDA_ * 2) + q * 16,
                    A + ((size_t)(bm + row) * K + kk));
      async_ld_b128(wsB + bsel * TILEB + row * (LDA_ * 2) + q * 16,
                    W + ((size_t)(bn + row) * K + kk));
    }
  };
  auto compute = [&](int bsel) {
    const unsigned short* Ab = As + bsel * TILEHB;
    const unsigned short* Wb = Ws + bsel * TILEHB;
    const int ar  = lane & 15;
    const int akh = (lane >> 4) * 8;
    const int bkh = (lane >> 4) * 16;
    v16bf af[2], bfr[4];
#pragma unroll
    for (int i = 0; i < 2; ++i)
      af[i] = ld_frag16(&Ab[(wm*32 + i*16 + ar) * LDA_ + akh], 16);
#pragma unroll
    for (int j = 0; j < 4; ++j)
      bfr[j] = ld_frag16(&Wb[(wn*64 + j*16 + ar) * LDA_ + akh], 16);
#pragma unroll
    for (int i = 0; i < 2; ++i)
#pragma unroll
      for (int j = 0; j < 4; ++j)
        acc[i][j] = __builtin_amdgcn_wmma_f32_16x16x32_bf16(
            false, af[i], false, bfr[j], (short)0, acc[i][j], false, false);
  };

  stage(0, 0);
  stage(1, 1);
  for (int s = 0; s < nk - 1; ++s) {
    wait_async4();                    // retire older stage, keep newer in flight
    __syncthreads();
    compute(s & 1);
    __syncthreads();                  // all waves done reading buf (s&1)
    if (s + 2 < nk) stage(s + 2, s & 1);
  }
  wait_async0();                      // peeled tail
  __syncthreads();
  compute((nk - 1) & 1);

  // epilogue: C/D layout: vgpr r of lane -> row (lane>=16 ? 8:0)+r, col lane&15
  const int mrow0 = bm + wm * 32 + (lane >> 4) * 8;
#pragma unroll
  for (int i = 0; i < 2; ++i) {
#pragma unroll
    for (int j = 0; j < 4; ++j) {
      const int col  = bn + wn * 64 + j * 16 + (lane & 15);
      const float bc = bias[col];
#pragma unroll
      for (int r = 0; r < 8; ++r) {
        float v = acc[i][j][r] + bc;
        if (GELU) v = gelu_fast(v);
        const size_t idx = (size_t)(mrow0 + i * 16 + r) * (size_t)Ncols + col;
        if (OUT_BF16) reinterpret_cast<unsigned short*>(Out)[idx] = f2bf(v);
        else          reinterpret_cast<float*>(Out)[idx] = v + resid[idx];
      }
    }
  }
}

// ---------------- launch ----------------

extern "C" void kernel_launch(void* const* d_in, const int* in_sizes, int n_in,
                              void* d_out, int out_size, void* d_ws, size_t ws_size,
                              hipStream_t stream) {
  const float* x    = (const float*)d_in[0];
  const float* n1w  = (const float*)d_in[2];
  const float* n1b  = (const float*)d_in[3];
  const float* klnw = (const float*)d_in[4];
  const float* klnb = (const float*)d_in[5];
  const float* vlnw = (const float*)d_in[6];
  const float* vlnb = (const float*)d_in[7];
  const float* n2w  = (const float*)d_in[8];
  const float* n2b  = (const float*)d_in[9];
  const float* fc1w = (const float*)d_in[10];
  const float* fc1b = (const float*)d_in[11];
  const float* fc2w = (const float*)d_in[12];
  const float* fc2b = (const float*)d_in[13];
  float* out = (float*)d_out;

  char* p = (char*)d_ws;
  float*          h   = (float*)p;          p += (size_t)M_ * C_ * 4;     // 134 MB
  unsigned short* qbf = (unsigned short*)p; p += (size_t)M_ * C_ * 2;     //  67 MB
  unsigned short* kbf = (unsigned short*)p; p += (size_t)M_ * C_ * 2;     //  67 MB
  unsigned short* vbf = (unsigned short*)p; p += (size_t)M_ * C_ * 2;     //  67 MB
  float*          kv  = (float*)p;          p += (size_t)B_*H_*D_*D_ * 4; // 256 KB
  unsigned short* mbf = (unsigned short*)p; p += (size_t)M_ * C_ * 2;     //  67 MB
  unsigned short* act = (unsigned short*)p; p += (size_t)M_ * HID_ * 2;   // 268 MB
  unsigned short* w1  = (unsigned short*)p; p += (size_t)HID_ * C_ * 2;   //   2 MB
  unsigned short* w2  = (unsigned short*)p;                               //   2 MB

  cvt_f32_bf16<<<(HID_ * C_) / 1024, 256, 0, stream>>>(fc1w, w1);
  cvt_f32_bf16<<<(HID_ * C_) / 1024, 256, 0, stream>>>(fc2w, w2);
  hipMemsetAsync(kv, 0, (size_t)B_ * H_ * D_ * D_ * 4, stream);

  ln1_qkv_kernel<<<M_ / 8, 256, 0, stream>>>(x, n1w, n1b, klnw, klnb, vlnw, vlnb,
                                             h, qbf, kbf, vbf);
  kv_accum_kernel<<<dim3(N_ / 1024, B_ * H_), 256, 0, stream>>>(kbf, vbf, kv);
  attn_resid_kernel<<<M_ / 16, 256, 0, stream>>>(x, h, qbf, kv, out);
  ln2_kernel<<<M_ / 8, 256, 0, stream>>>(out, n2w, n2b, mbf);

  // grid.x = N tiles (fast), grid.y = M tiles -> A rows reused across consecutive blocks
  gemm_bf16_kernel<true,  true ><<<dim3(HID_ / 128, M_ / 128), 256, 0, stream>>>(
      mbf, w1, fc1b, nullptr, act, C_, HID_);
  gemm_bf16_kernel<false, false><<<dim3(C_ / 128, M_ / 128), 256, 0, stream>>>(
      act, w2, fc2b, out, out, HID_, C_);
}
